// OrthoLinear_83167746720183
// MI455X (gfx1250) — matrix operand order repaired and checked by reading, not compiled, not run on hardware
//
#include <hip/hip_runtime.h>

typedef __attribute__((ext_vector_type(16))) _Float16 v16h;
typedef __attribute__((ext_vector_type(8)))  _Float16 v8h;
typedef __attribute__((ext_vector_type(8)))  float    v8f;
typedef unsigned int u32x4 __attribute__((ext_vector_type(4)));
typedef int          i32x8 __attribute__((ext_vector_type(8)));
typedef int          i32x4 __attribute__((ext_vector_type(4)));

#define TOKENS 512
#define IN_F   8192
#define OUT_F  8192
#define TM 128
#define TN 128
#define TK 64
#define BPAD 16  // halves of pad -> Bs row stride 288B (32B multiple)

#if defined(__has_builtin)
#if __has_builtin(__builtin_amdgcn_tensor_load_to_lds)
#define HAVE_TDM 1
#endif
#endif
#ifndef HAVE_TDM
#define HAVE_TDM 0
#endif

// ---------------- prepass: x f32 -> f16 (8 MB, one-time) ----------------
__global__ __launch_bounds__(256) void x_to_f16(const float* __restrict__ x,
                                                _Float16* __restrict__ xh) {
    size_t i = ((size_t)blockIdx.x * 256 + threadIdx.x) * 8;
    float4 a = ((const float4*)(x + i))[0];
    float4 b = ((const float4*)(x + i))[1];
    v8h h;
    h[0] = (_Float16)a.x; h[1] = (_Float16)a.y;
    h[2] = (_Float16)a.z; h[3] = (_Float16)a.w;
    h[4] = (_Float16)b.x; h[5] = (_Float16)b.y;
    h[6] = (_Float16)b.z; h[7] = (_Float16)b.w;
    *(v8h*)(xh + i) = h;
}

// ---------------- dense int4-dequant GEMM via WMMA (+TDM x-tiles) --------
template <bool USE_TDM>
__global__ __launch_bounds__(256) void ortho_gemm_wmma(
    const float*    __restrict__ x,       // [TOKENS][IN_F] f32
    const _Float16* __restrict__ xh,      // [TOKENS][IN_F] f16 (TDM path)
    const int*      __restrict__ packed,  // [OUT_F][IN_F/2] int32, byte-valued
    const float*    __restrict__ scales,  // [OUT_F]
    float*          __restrict__ out)     // [TOKENS][OUT_F]
{
    __shared__ __align__(32) _Float16 As[TM][TK];        // 16 KB, contiguous
    __shared__ __align__(32) _Float16 Bs[TK][TN + BPAD]; // 18 KB (transposed W)

    const int tid  = threadIdx.x;
    const int wave = tid >> 5;
    const int lane = tid & 31;
    const int m0 = blockIdx.y * TM;     // token base
    const int n0 = blockIdx.x * TN;     // out-feature base

    // 2-D wave split: 2 M-groups x 4 N-groups
    const int mgroup = wave >> 2;       // 0..1 -> M-subtiles mgroup*4 .. +3
    const int ngroup = wave & 3;        // 0..3 -> columns ngroup*32 .. +31
    const int nbase  = ngroup * 32;

    const int L  = lane & 15;
    const int hi = lane >> 4;

    v8f acc[4][2];
#pragma unroll
    for (int mt = 0; mt < 4; ++mt)
#pragma unroll
        for (int nt = 0; nt < 2; ++nt)
#pragma unroll
            for (int r = 0; r < 8; ++r) acc[mt][nt][r] = 0.0f;

    for (int k0 = 0; k0 < IN_F; k0 += TK) {
#if HAVE_TDM
        if (USE_TDM) {
            // ---- TDM: DMA x f16 tile [TM rows x TK cols] into As ----
            if (wave == 0) {
                unsigned long long ga =
                    (unsigned long long)(const void*)(xh + (size_t)m0 * IN_F + k0);
                u32x4 g0;
                g0.x = 1u;                                    // count=1, user D#
                g0.y = (unsigned int)(unsigned long long)(void*)&As[0][0]; // LDS byte addr
                g0.z = (unsigned int)(ga & 0xFFFFFFFFull);    // global_addr[31:0]
                g0.w = (unsigned int)((ga >> 32) & 0x1FFFFFFull) | (2u << 30); // [56:32]|type=2
                i32x8 g1;
                g1[0] = 1 << 16;                              // data_size=1 (2 bytes)
                g1[1] = (int)((IN_F & 0xFFFF) << 16);         // tensor_dim0 lo16
                g1[2] = (int)(((IN_F >> 16) & 0xFFFF) |       // tensor_dim0 hi16
                              ((TOKENS & 0xFFFF) << 16));     // tensor_dim1 lo16
                g1[3] = (int)(TK << 16);                      // tile_dim0 = 64
                g1[4] = TM;                                   // tile_dim1 = 128
                g1[5] = IN_F;                                 // dim0_stride lo32
                g1[6] = 0;                                    // stride hi / dim1_stride
                g1[7] = 0;
                i32x4 z4 = {0, 0, 0, 0};
#if __clang_major__ >= 23
                i32x8 z8 = {0, 0, 0, 0, 0, 0, 0, 0};
                __builtin_amdgcn_tensor_load_to_lds(g0, g1, z4, z4, z8, 0);
#else
                __builtin_amdgcn_tensor_load_to_lds(g0, g1, z4, z4, 0);
#endif
            }
        } else
#endif
        {
            // ---- fallback: stage x tile f32 -> f16 by VALU ----
#pragma unroll
            for (int i = 0; i < (TM * TK / 4) / 256; ++i) {   // 8 iters
                int idx = tid + 256 * i;
                int r   = idx >> 4;        // TK/4 = 16 float4 per row
                int c4  = idx & 15;
                const float4 v =
                    ((const float4*)(x + (size_t)(m0 + r) * IN_F + k0))[c4];
                int c = c4 << 2;
                As[r][c + 0] = (_Float16)v.x;
                As[r][c + 1] = (_Float16)v.y;
                As[r][c + 2] = (_Float16)v.z;
                As[r][c + 3] = (_Float16)v.w;
            }
        }

        // ---- stage W tile: dequant int4 -> f16, transposed Bs[k][n] ----
#pragma unroll
        for (int i = 0; i < (TN * (TK / 2)) / 256; ++i) { // 16 iters
            int idx = tid + 256 * i;
            int row = idx >> 5;        // TK/2 = 32 packed ints per row
            int j   = idx & 31;
            int p   = packed[(size_t)(n0 + row) * (IN_F / 2) + (k0 >> 1) + j];
            float s = scales[n0 + row];
            Bs[2 * j + 0][row] = (_Float16)((float)((p       & 0xF) - 8) * s);
            Bs[2 * j + 1][row] = (_Float16)((float)(((p >> 4) & 0xF) - 8) * s);
        }
        if (k0 + TK < IN_F) {  // pull next W tile toward the caches
            __builtin_prefetch(packed + (size_t)(n0 + (tid >> 1)) * (IN_F / 2) +
                                   ((k0 + TK) >> 1), 0, 0);
        }
#if HAVE_TDM
        if (USE_TDM && wave == 0) __builtin_amdgcn_s_wait_tensorcnt((short)0);
#endif
        __syncthreads();

#pragma unroll
        for (int kk = 0; kk < 2; ++kk) {
            const int ak = kk * 32 + hi * 8;

            // 2 B fragments: column N = L, K rows in the same order as A (ak+e, ak+16+e)
            v16h bf[2];
#pragma unroll
            for (int nt = 0; nt < 2; ++nt) {
#pragma unroll
                for (int e = 0; e < 8; ++e) {
                    bf[nt][e] = Bs[ak + e][nbase + nt * 16 + L];
                    bf[nt][e + 8] = Bs[ak + 16 + e][nbase + nt * 16 + L];
                }
            }
            // 4 A fragments (this wave's M-group)
            v16h af[4];
#pragma unroll
            for (int mt = 0; mt < 4; ++mt) {
                int am = (mgroup * 4 + mt) * 16 + L;
                v8h a_lo = *(const v8h*)&As[am][ak];
                v8h a_hi = *(const v8h*)&As[am][ak + 16];
#pragma unroll
                for (int e = 0; e < 8; ++e) {
                    af[mt][e] = a_lo[e];
                    af[mt][e + 8] = a_hi[e];
                }
            }
            // 8 WMMAs reusing af x2 and bf x4 (8 b128 A loads, 32 16-bit B loads / 8 wmma)
#pragma unroll
            for (int mt = 0; mt < 4; ++mt)
#pragma unroll
                for (int nt = 0; nt < 2; ++nt) {
                    acc[mt][nt] = __builtin_amdgcn_wmma_f32_16x16x32_f16(
                        false, af[mt], false, bf[nt], (short)0, acc[mt][nt],
                        false, false);
                }
        }
        __syncthreads();
    }

    // ---- epilogue: C layout => VGPR r holds M=r (lanes 0-15) / M=8+r ----
#pragma unroll
    for (int mt = 0; mt < 4; ++mt) {
#pragma unroll
        for (int nt = 0; nt < 2; ++nt) {
#pragma unroll
            for (int r = 0; r < 8; ++r) {
                int m = m0 + (mgroup * 4 + mt) * 16 + hi * 8 + r;
                int n = n0 + nbase + nt * 16 + L;
                out[(size_t)m * OUT_F + n] = acc[mt][nt][r];
            }
        }
    }
}

// ---------------- CSR residual: out[t,o] += alpha * sum_j v*x[t,col] ----
__global__ __launch_bounds__(256) void ortho_residual(
    const float*    __restrict__ x,     // [TOKENS][IN_F]
    const _Float16* __restrict__ vals,  // [NNZ]
    const int*      __restrict__ cols,  // [NNZ]
    const int*      __restrict__ ptr,   // [OUT_F+1]
    const float*    __restrict__ alphap,
    float*          __restrict__ out)   // [TOKENS][OUT_F]
{
    const int o = blockIdx.x;
    const int start = ptr[o];
    const int nnz   = ptr[o + 1] - start;   // 409 in this problem

    __shared__ int   scol[512];
    __shared__ float sval[512];
    for (int i = threadIdx.x; i < nnz; i += 256) {
        scol[i] = cols[start + i];
        sval[i] = (float)vals[start + i];
    }
    __syncthreads();

    const float alpha = alphap[0];
    for (int t = threadIdx.x; t < TOKENS; t += 256) {
        const float* xr = x + (size_t)t * IN_F;
        float s = 0.0f;
        for (int i = 0; i < nnz; ++i) s += sval[i] * xr[scol[i]];
        out[(size_t)t * OUT_F + o] += alpha * s;
    }
}

extern "C" void kernel_launch(void* const* d_in, const int* in_sizes, int n_in,
                              void* d_out, int out_size, void* d_ws, size_t ws_size,
                              hipStream_t stream) {
    (void)in_sizes; (void)n_in; (void)out_size;
    const float*    x      = (const float*)d_in[0];
    const int*      packed = (const int*)d_in[1];
    const float*    scales = (const float*)d_in[2];
    const _Float16* vals   = (const _Float16*)d_in[3];
    const int*      cols   = (const int*)d_in[4];
    const int*      ptr    = (const int*)d_in[5];
    const float*    alpha  = (const float*)d_in[6];
    float* out = (float*)d_out;

    dim3 grid(OUT_F / TN, TOKENS / TM);  // (64, 4)

    const size_t xh_bytes = (size_t)TOKENS * IN_F * sizeof(_Float16);
    if (HAVE_TDM && ws_size >= xh_bytes) {
        _Float16* xh = (_Float16*)d_ws;
        x_to_f16<<<(TOKENS * IN_F) / (256 * 8), 256, 0, stream>>>(x, xh);
        ortho_gemm_wmma<true><<<grid, 256, 0, stream>>>(x, xh, packed, scales, out);
    } else {
        ortho_gemm_wmma<false><<<grid, 256, 0, stream>>>(x, nullptr, packed, scales, out);
    }
    ortho_residual<<<OUT_F, 256, 0, stream>>>(x, vals, cols, ptr, alpha, out);
}
